// diff_pred_26319559589961
// MI455X (gfx1250) — matrix operand order repaired and checked by reading, not compiled, not run on hardware
//
#include <hip/hip_runtime.h>

typedef __attribute__((ext_vector_type(16))) _Float16 v16h;
typedef __attribute__((ext_vector_type(8)))  float    v8f;

#define FRAME   160
#define NTAP    16
#define T_LEN   2400
#define NFRAME  15
#define WPB     8   // waves per block

// mu-law (0..255) -> linear.  exp(|u|/128 * ln256) == exp2(|u| * 8/128)
__device__ __forceinline__ float u2l(float u) {
    float un = u - 128.0f;
    float r  = (32768.0f / 255.0f) * (exp2f(fabsf(un) * 0.0625f) - 1.0f);
    return copysignf(r, un);
}

// linear -> mu-law (0..255).  128*log1p(s*|x|)/ln256 == 16*log2(1 + s*|x|)
__device__ __forceinline__ float l2u(float x) {
    float u = 16.0f * log2f(fmaf(255.0f / 32768.0f, fabsf(x), 1.0f));
    float r = 128.0f + copysignf(u, x);
    return fminf(fmaxf(r, 0.0f), 255.0f);
}

__global__ __launch_bounds__(256) void lpc_pred_wmma_kernel(
    const float* __restrict__ sig,   // (B, 2400) mu-law samples, f32
    const float* __restrict__ lpc,   // (B, 15, 16) f32
    float* __restrict__ out,         // (B, 2400) f32
    int total_waves)
{
    __shared__ _Float16 sx[WPB][192];  // decoded signal window (f16)
    __shared__ _Float16 sc[WPB][48];   // zero-padded coeffs: sc[16+v] = c[v]

    const int lane = threadIdx.x & 31;
    const int w    = threadIdx.x >> 5;
    const int gw   = blockIdx.x * WPB + w;
    const bool valid = (gw < total_waves);
    const int gwc  = valid ? gw : (total_waves - 1);

    const int b  = gwc / NFRAME;
    const int f  = gwc - b * NFRAME;
    const int t0 = f * FRAME;

    // ---- stage u-law-decoded signal window t in [t0-17, t0+174] as f16 ----
    const float* sb = sig + (size_t)b * T_LEN;
    #pragma unroll
    for (int it = 0; it < 6; ++it) {
        int idx = it * 32 + lane;          // 6*32 = 192, full fill (no NaNs)
        int t   = t0 - 17 + idx;
        float v = 0.0f;
        if (t >= 0 && t < T_LEN) v = u2l(sb[t]);
        sx[w][idx] = (_Float16)v;
    }
    // ---- stage zero-padded coefficients: sc[j] = c[j-16] for j in 16..31 ----
    const float* cb = lpc + ((size_t)b * NFRAME + f) * NTAP;
    #pragma unroll
    for (int it = 0; it < 2; ++it) {
        int idx = it * 32 + lane;
        if (idx < 48) {
            int v   = idx - 16;
            float c = (v >= 0 && v < NTAP) ? cb[v] : 0.0f;
            sc[w][idx] = (_Float16)c;
        }
    }
    __syncthreads();   // uniform across block (clamped gw keeps flow uniform)

    const int hi = lane >> 4;     // 0: lanes 0-15, 1: lanes 16-31
    const int mn = lane & 15;     // A: row m; B/D: column n

    // ---- A fragment: Toeplitz coeffs, A[m,K] = cpad[31 + m - K] ----
    // f16 A layout: lanes 0-15 halfs 0-7 => K 0-7, halfs 8-15 => K 16-23;
    //               lanes 16-31: +8.
    v16h a;
    #pragma unroll
    for (int h = 0; h < 16; ++h) {
        int K = ((h < 8) ? h : h + 8) + hi * 8;
        a[h] = sc[w][31 + mn - K];            // indices 0..46, zeros pad band
    }

    // ---- B fragment: B[k,n] = xt[t0 + 16n + k - 15], lds pos = 16n + k + 2
    // f16 B layout: lanes 0-15 halfs 0-7 => K 0-7, halfs 8-15 => K 16-23; lanes 16-31: +8.
    v16h bm;
    #pragma unroll
    for (int h = 0; h < 16; ++h) {
        int k = ((h < 8) ? h : h + 8) + hi * 8;
        _Float16 v = (_Float16)0.0f;
        if (mn < 10) v = sx[w][mn * 16 + k + 2];  // max pos 177 < 192
        bm[h] = v;
    }

    // ---- one WMMA computes the whole 160-sample frame ----
    v8f acc = {};
    acc = __builtin_amdgcn_wmma_f32_16x16x32_f16(
        /*neg_a=*/false, a, /*neg_b=*/false, bm,
        /*c_mod=*/(short)0, acc, /*reuse_a=*/false, /*reuse_b=*/false);

    // ---- D layout: VGPR r -> M = r + 8*hi, N = mn.  y = l2u(-pred) ----
    if (valid && mn < 10) {
        float* ob = out + (size_t)b * T_LEN + t0 + mn * 16 + hi * 8;
        float4 lo, hh;
        lo.x = l2u(-acc[0]); lo.y = l2u(-acc[1]);
        lo.z = l2u(-acc[2]); lo.w = l2u(-acc[3]);
        hh.x = l2u(-acc[4]); hh.y = l2u(-acc[5]);
        hh.z = l2u(-acc[6]); hh.w = l2u(-acc[7]);
        *(float4*)(ob)     = lo;   // 32B-aligned: t offset is a multiple of 8
        *(float4*)(ob + 4) = hh;
    }
}

extern "C" void kernel_launch(void* const* d_in, const int* in_sizes, int n_in,
                              void* d_out, int out_size, void* d_ws, size_t ws_size,
                              hipStream_t stream) {
    (void)n_in; (void)out_size; (void)d_ws; (void)ws_size;
    const float* sig = (const float*)d_in[0];
    const float* lpc = (const float*)d_in[1];
    float* out = (float*)d_out;

    const int Bn          = in_sizes[1] / (NFRAME * NTAP);  // 1024
    const int total_waves = Bn * NFRAME;                    // 15360
    const int blocks      = (total_waves + WPB - 1) / WPB;  // 1920

    lpc_pred_wmma_kernel<<<blocks, 256, 0, stream>>>(sig, lpc, out, total_waves);
}
